// PosNet_20985210208672
// MI455X (gfx1250) — matrix operand order repaired and checked by reading, not compiled, not run on hardware
//
#include <hip/hip_runtime.h>

typedef __attribute__((ext_vector_type(16))) __bf16 v16bf;
typedef __attribute__((ext_vector_type(8)))  __bf16 v8bf;
typedef __attribute__((ext_vector_type(8)))  float  v8f;

#define LEAKY 0.01f
#define BNEPS 1e-5f

// ---------------- prep kernels ----------------
__global__ void k_zero_f32(float* __restrict__ p, long long n) {
  long long i = blockIdx.x * (long long)blockDim.x + threadIdx.x;
  if (i < n) p[i] = 0.f;
}

__global__ void k_deg_count(const int* __restrict__ col, float* __restrict__ deg, int E) {
  int e = blockIdx.x * blockDim.x + threadIdx.x;
  if (e < E) atomicAdd(&deg[col[e]], 1.f);
}

__global__ void k_dinv(const float* __restrict__ deg, float* __restrict__ dinv,
                       float* __restrict__ dinv2, int N) {
  int v = blockIdx.x * blockDim.x + threadIdx.x;
  if (v < N) {
    float r = rsqrtf(deg[v] + 1.f);  // +1 for self loop
    dinv[v]  = r;
    dinv2[v] = r * r;
  }
}

__global__ void k_norm(const int* __restrict__ row, const int* __restrict__ col,
                       const float* __restrict__ dinv, float* __restrict__ nrm, int E) {
  int e = blockIdx.x * blockDim.x + threadIdx.x;
  if (e < E) nrm[e] = dinv[row[e]] * dinv[col[e]];
}

__global__ void k_cvt_z1(const float* __restrict__ z1, __bf16* __restrict__ xb, int N) {
  // [N,16] f32 -> [N,32] bf16 zero-padded (K padded to 32 for WMMA)
  long long i = blockIdx.x * (long long)blockDim.x + threadIdx.x;
  if (i >= (long long)N * 32) return;
  int v = (int)(i >> 5), f = (int)(i & 31);
  xb[i] = (f < 16) ? (__bf16)z1[v * 16 + f] : (__bf16)0.f;
}

// Convert W [Fin,Fout] f32 -> W^T [Fout,Kpad] bf16 (zero K-padding).
// Transposed layout makes the per-block W tile a CONTIGUOUS region so the GEMM
// can stage it with global_load_async_to_lds_b128.
__global__ void k_cvt_wT(const float* __restrict__ W, __bf16* __restrict__ WbT,
                         int Fin, int Kpad, int Fout) {
  long long i = blockIdx.x * (long long)blockDim.x + threadIdx.x;
  if (i >= (long long)Kpad * Fout) return;
  int c = (int)(i / Kpad), k = (int)(i - (long long)c * Kpad);
  WbT[i] = (k < Fin) ? (__bf16)W[(long long)k * Fout + c] : (__bf16)0.f;
}

// ---------------- WMMA GEMM: H = Xbf16 @ Wbf16 (f32 accum) ----------------
// Block tile: 128 rows x (CT*16) cols. 8 waves; wave w owns rows 16w..16w+15 and
// keeps CT accumulators across the column tiles (A fragment reused CT times).
// grid.x = ceil(N/128), grid.y = Fout/(16*CT), 256 threads.
template <int KPAD, int CT>
__global__ __launch_bounds__(256)
void k_gemm_bf16(const __bf16* __restrict__ X, const __bf16* __restrict__ WbT,
                 float* __restrict__ Hout, int N, int Fout) {
  __shared__ __bf16 wlds[CT * 16 * KPAD];  // W^T tile: [col][k], contiguous rows
  const int tid  = threadIdx.x;
  const int wave = tid >> 5, lane = tid & 31;
  const int hk = lane >> 4, l15 = lane & 15;
  const int colBase = blockIdx.y * (CT * 16);

  // ---- async stage of the contiguous W^T tile into LDS ----
  {
    const int bytesTotal = CT * 16 * KPAD * 2;
    const char* gsrc = (const char*)(WbT + (long long)colBase * KPAD);
    unsigned ldsBase = (unsigned)(size_t)(void*)wlds;  // low 32 bits = LDS offset
    for (int off = tid * 16; off < bytesTotal; off += 256 * 16) {
      unsigned laddr = ldsBase + (unsigned)off;
      const char* g = gsrc + off;
      asm volatile("global_load_async_to_lds_b128 %0, %1, off"
                   :: "v"(laddr), "v"(g) : "memory");
    }
    asm volatile("s_wait_asynccnt 0x0" ::: "memory");
  }
  __syncthreads();

  const int rowBase = (blockIdx.x << 7) + (wave << 4);
  if (rowBase >= N) return;  // wave-uniform: EXEC stays all-1s for WMMA

  const __bf16* xrow = X + (long long)(rowBase + l15) * KPAD;
  v8f acc[CT];
#pragma unroll
  for (int t = 0; t < CT; ++t) acc[t] = (v8f){0.f, 0.f, 0.f, 0.f, 0.f, 0.f, 0.f, 0.f};

#pragma unroll
  for (int k0 = 0; k0 < KPAD; k0 += 32) {
    __builtin_prefetch((const void*)(xrow + k0 + 128), 0, 0);  // global_prefetch_b8
    // A (16x32 bf16): lane<16 holds K={0..7,16..23}, lane>=16 holds K={8..15,24..31}
    v8bf alo = *(const v8bf*)(xrow + k0 + (hk << 3));
    v8bf ahi = *(const v8bf*)(xrow + k0 + 16 + (hk << 3));
    v16bf a = __builtin_shufflevector(alo, ahi, 0,1,2,3,4,5,6,7,8,9,10,11,12,13,14,15);
#pragma unroll
    for (int t = 0; t < CT; ++t) {
      // B (32x16 bf16): lanes 0-15 hold K={0..7,16..23} of col=lane, lanes 16-31 K={8..15,24..31}
      const __bf16* wc = wlds + (t * 16 + l15) * KPAD + (hk << 3);
      v8bf blo = *(const v8bf*)(wc + k0);
      v8bf bhi = *(const v8bf*)(wc + k0 + 16);
      v16bf b = __builtin_shufflevector(blo, bhi,
                                        0,1,2,3,4,5,6,7,8,9,10,11,12,13,14,15);
      acc[t] = __builtin_amdgcn_wmma_f32_16x16x32_bf16(false, a, false, b, (short)0,
                                                       acc[t], false, false);
    }
  }
  // C/D layout: VGPR d -> M = d + 8*hk, N = l15
#pragma unroll
  for (int t = 0; t < CT; ++t) {
    float* hp = Hout + (long long)(rowBase + (hk << 3)) * Fout + colBase + t * 16 + l15;
#pragma unroll
    for (int d = 0; d < 8; ++d) hp[(long long)d * Fout] = acc[t][d];
  }
}

// ---------------- aggregation ----------------
__global__ void k_init_selfloop(const float* __restrict__ Hin, const float* __restrict__ bias,
                                const float* __restrict__ dinv2, float* __restrict__ Y,
                                int N, int Fout) {
  long long i = blockIdx.x * (long long)blockDim.x + threadIdx.x;
  if (i >= (long long)N * Fout) return;
  int v = (int)(i / Fout), f = (int)(i - (long long)v * Fout);
  Y[i] = bias[f] + Hin[i] * dinv2[v];
}

__global__ void k_scatter(const float* __restrict__ Hin, float* __restrict__ Y,
                          const int* __restrict__ row, const int* __restrict__ col,
                          const float* __restrict__ nrm, int E, int Fout) {
  const int chunks = Fout >> 2;
  long long gid = blockIdx.x * (long long)blockDim.x + threadIdx.x;
  if (gid >= (long long)E * chunks) return;
  int e = (int)(gid / chunks);
  int f = (int)(gid - (long long)e * chunks) << 2;
  float w = nrm[e];
  const float4 h = *(const float4*)(Hin + (long long)row[e] * Fout + f);
  float* y = Y + (long long)col[e] * Fout + f;
  atomicAdd(y + 0, h.x * w);
  atomicAdd(y + 1, h.y * w);
  atomicAdd(y + 2, h.z * w);
  atomicAdd(y + 3, h.w * w);
}

// ---------------- batch norm ----------------
__global__ void k_bn_stats(const float* __restrict__ Y, float* __restrict__ sums,
                           int N, int Fout, int rowsPerBlock) {
  int r0 = blockIdx.x * rowsPerBlock;
  int r1 = r0 + rowsPerBlock; if (r1 > N) r1 = N;
  for (int f = threadIdx.x; f < Fout; f += blockDim.x) {
    float s = 0.f, ss = 0.f;
    for (int r = r0; r < r1; ++r) {
      float x = Y[(long long)r * Fout + f];
      s += x; ss += x * x;
    }
    atomicAdd(&sums[f], s);
    atomicAdd(&sums[Fout + f], ss);
  }
}

__global__ void k_bn_apply(const float* __restrict__ Y, const float* __restrict__ sums,
                           const float* __restrict__ g, const float* __restrict__ be,
                           __bf16* __restrict__ Xout, int N, int Fout) {
  long long i = blockIdx.x * (long long)blockDim.x + threadIdx.x;
  if (i >= (long long)N * Fout) return;
  int f = (int)(i - (i / Fout) * (long long)Fout);
  float invn = 1.f / (float)N;
  float mean = sums[f] * invn;
  float var  = sums[Fout + f] * invn - mean * mean;
  float sc = g[f] * rsqrtf(var + BNEPS);
  float x = (Y[i] - mean) * sc + be[f];
  x = (x >= 0.f) ? x : LEAKY * x;
  Xout[i] = (__bf16)x;  // next layer's WMMA A operand
}

// ---------------- final MLP + residual ----------------
__global__ void k_mlp(const __bf16* __restrict__ X, const float* __restrict__ lw1,
                      const float* __restrict__ lb1, const float* __restrict__ lw2,
                      const float* __restrict__ lb2, const float* __restrict__ xpos,
                      float* __restrict__ out, int N) {
  int v = blockIdx.x * blockDim.x + threadIdx.x;
  if (v >= N) return;
  float xin[32];
  const __bf16* xr = X + (long long)v * 32;
#pragma unroll
  for (int k = 0; k < 32; ++k) xin[k] = (float)xr[k];
  float h[16];
#pragma unroll
  for (int j = 0; j < 16; ++j) {
    float a = lb1[j];
    for (int k = 0; k < 32; ++k) a += xin[k] * lw1[k * 16 + j];
    h[j] = (a >= 0.f) ? a : LEAKY * a;
  }
#pragma unroll
  for (int j = 0; j < 3; ++j) {
    float a = lb2[j];
    for (int k = 0; k < 16; ++k) a += h[k] * lw2[k * 3 + j];
    out[(long long)v * 3 + j] = xpos[(long long)v * 3 + j] + a;
  }
}

// ---------------- host ----------------
static void launch_gemm(int Kpad, int CT, dim3 g, dim3 b, hipStream_t s,
                        const __bf16* X, const __bf16* WbT, float* H, int N, int Fout) {
  if (CT == 2) {
    switch (Kpad) {
      case 32: k_gemm_bf16<32, 2><<<g, b, 0, s>>>(X, WbT, H, N, Fout); break;
      case 64: k_gemm_bf16<64, 2><<<g, b, 0, s>>>(X, WbT, H, N, Fout); break;
    }
  } else {
    switch (Kpad) {
      case 32:  k_gemm_bf16<32, 4> <<<g, b, 0, s>>>(X, WbT, H, N, Fout); break;
      case 64:  k_gemm_bf16<64, 4> <<<g, b, 0, s>>>(X, WbT, H, N, Fout); break;
      case 128: k_gemm_bf16<128, 4><<<g, b, 0, s>>>(X, WbT, H, N, Fout); break;
      case 256: k_gemm_bf16<256, 4><<<g, b, 0, s>>>(X, WbT, H, N, Fout); break;
      case 512: k_gemm_bf16<512, 4><<<g, b, 0, s>>>(X, WbT, H, N, Fout); break;
    }
  }
}

extern "C" void kernel_launch(void* const* d_in, const int* in_sizes, int n_in,
                              void* d_out, int out_size, void* d_ws, size_t ws_size,
                              hipStream_t stream) {
  static const int HD[15] = {16,32,64,128,256,256,512,512,256,256,128,64,32,16,3};
  const float* z1   = (const float*)d_in[0];
  const float* xpos = (const float*)d_in[1];
  const int*   eidx = (const int*)d_in[2];
  const int N = in_sizes[0] / 16;
  const int E = in_sizes[2] / 2;
  const int* erow = eidx;
  const int* ecol = eidx + E;

  const float *Wp[12], *bp[12], *gp[12], *bep[12];
  for (int i = 0; i < 12; ++i) {
    Wp[i]  = (const float*)d_in[3 + 4 * i];
    bp[i]  = (const float*)d_in[4 + 4 * i];
    gp[i]  = (const float*)d_in[5 + 4 * i];
    bep[i] = (const float*)d_in[6 + 4 * i];
  }
  const float* lw1 = (const float*)d_in[51];
  const float* lb1 = (const float*)d_in[52];
  const float* lw2 = (const float*)d_in[53];
  const float* lb2 = (const float*)d_in[54];

  // workspace carve
  char* ws = (char*)d_ws;
  size_t off = 0;
  auto carve = [&](size_t bytes) -> char* {
    bytes = (bytes + 255) & ~(size_t)255;
    char* p = ws + off; off += bytes; return p;
  };
  __bf16* Xbf  = (__bf16*)carve((size_t)N * 512 * sizeof(__bf16));
  float*  Hbuf = (float*) carve((size_t)N * 512 * sizeof(float));
  float*  Ybuf = (float*) carve((size_t)N * 512 * sizeof(float));
  float*  deg  = (float*) carve((size_t)N * sizeof(float));
  float*  dinv = (float*) carve((size_t)N * sizeof(float));
  float*  dnv2 = (float*) carve((size_t)N * sizeof(float));
  float*  nrm  = (float*) carve((size_t)E * sizeof(float));
  float*  sums = (float*) carve(1024 * sizeof(float));
  __bf16* WbT[12];
  int Kpads[12], CTs[12];
  for (int li = 0; li < 12; ++li) {
    int Fin = HD[li], Fout = HD[li + 1];
    int Kpad = (Fin < 32) ? 32 : Fin;
    Kpads[li] = Kpad;
    CTs[li] = (Fout >= 64) ? 4 : 2;
    WbT[li] = (__bf16*)carve((size_t)Kpad * Fout * sizeof(__bf16));
  }

  dim3 B(256);
  auto gr = [](long long n) { return dim3((unsigned)((n + 255) / 256)); };

  // graph normalization prep
  k_zero_f32 <<<gr(N), B, 0, stream>>>(deg, N);
  k_deg_count<<<gr(E), B, 0, stream>>>(ecol, deg, E);
  k_dinv     <<<gr(N), B, 0, stream>>>(deg, dinv, dnv2, N);
  k_norm     <<<gr(E), B, 0, stream>>>(erow, ecol, dinv, nrm, E);

  // bf16 conversions (W transposed for contiguous async-LDS staging)
  k_cvt_z1<<<gr((long long)N * 32), B, 0, stream>>>(z1, Xbf, N);
  for (int li = 0; li < 12; ++li) {
    int Fin = HD[li], Fout = HD[li + 1];
    k_cvt_wT<<<gr((long long)Kpads[li] * Fout), B, 0, stream>>>(Wp[li], WbT[li], Fin,
                                                                Kpads[li], Fout);
  }

  // 12 GCN layers
  for (int li = 0; li < 12; ++li) {
    int Fout = HD[li + 1];
    dim3 gg((unsigned)((N + 127) / 128), (unsigned)(Fout / (16 * CTs[li])));
    launch_gemm(Kpads[li], CTs[li], gg, B, stream, Xbf, WbT[li], Hbuf, N, Fout);
    k_init_selfloop<<<gr((long long)N * Fout), B, 0, stream>>>(Hbuf, bp[li], dnv2,
                                                               Ybuf, N, Fout);
    k_scatter<<<gr((long long)E * (Fout / 4)), B, 0, stream>>>(Hbuf, Ybuf, erow, ecol,
                                                               nrm, E, Fout);
    k_zero_f32<<<dim3(4), B, 0, stream>>>(sums, 2 * Fout);
    k_bn_stats<<<dim3((unsigned)((N + 511) / 512)), B, 0, stream>>>(Ybuf, sums, N,
                                                                    Fout, 512);
    k_bn_apply<<<gr((long long)N * Fout), B, 0, stream>>>(Ybuf, sums, gp[li], bep[li],
                                                          Xbf, N, Fout);
  }

  // final MLP + residual
  k_mlp<<<gr(N), B, 0, stream>>>(Xbf, lw1, lb1, lw2, lb2, xpos, (float*)d_out, N);
}
